// LSS_CAN_Mamba_60000693125961
// MI455X (gfx1250) — matrix operand order repaired and checked by reading, compiled, not run on hardware
//
#include <hip/hip_runtime.h>

// ---------------- constants ----------------
#define Bb 32
#define LL 2048
#define LP 2050   // padded length (1 halo row each side) for conv im2col

typedef __bf16 bf16_t;
typedef __attribute__((ext_vector_type(16))) __bf16 v16bf;
typedef __attribute__((ext_vector_type(8)))  __bf16 v8bf;
typedef __attribute__((ext_vector_type(8)))  float  v8f;

__device__ __forceinline__ v16bf cat8(v8bf lo, v8bf hi) {
  return __builtin_shufflevector(lo, hi, 0, 1, 2, 3, 4, 5, 6, 7, 8, 9, 10, 11, 12, 13, 14, 15);
}
__device__ __forceinline__ float silu_f(float x) { return x / (1.f + expf(-x)); }

// ---------------------------------------------------------------------------
// Register-resident GEMM: A row-major [M][K] bf16, BT transposed [N][K] bf16.
// Wave tile 64x32 (4x2 wmma tiles), block 8 waves -> 128x128, no LDS/barriers.
// Double-buffered K loop: fragments for step k+1 load while step k's WMMAs issue.
// ---------------------------------------------------------------------------
#define MW 4
#define NW 2

__device__ __forceinline__ void load_frags(const bf16_t* const (&pa)[MW],
                                           const bf16_t* const (&pb)[NW], int kk,
                                           v16bf (&a)[MW], v16bf (&b)[NW]) {
#pragma unroll
  for (int mi = 0; mi < MW; ++mi) {
    const bf16_t* p = pa[mi] + kk;
    a[mi] = cat8(*(const v8bf*)p, *(const v8bf*)(p + 16));
  }
#pragma unroll
  for (int ni = 0; ni < NW; ++ni) {
    const bf16_t* p = pb[ni] + kk;
    b[ni] = cat8(*(const v8bf*)p, *(const v8bf*)(p + 16));
  }
}

__device__ __forceinline__ void wmma8(const v16bf (&a)[MW], const v16bf (&b)[NW],
                                      v8f (&acc)[MW][NW]) {
#pragma unroll
  for (int mi = 0; mi < MW; ++mi)
#pragma unroll
    for (int ni = 0; ni < NW; ++ni)
      acc[mi][ni] = __builtin_amdgcn_wmma_f32_16x16x32_bf16(false, a[mi], false, b[ni],
                                                            (short)0, acc[mi][ni], false, false);
}

// EPI: 0 = inproj split (xm_pre | silu(z)), 1 = xproj (store ld=80, n<80), 2 = plain ld=256
template <int EPI>
__global__ void __launch_bounds__(256) k_gemm(const bf16_t* __restrict__ A, int lda,
                                              const bf16_t* __restrict__ BT, int ldb, int K,
                                              float* __restrict__ out0, float* __restrict__ out1) {
  int tid = threadIdx.x, lane = tid & 31, w = tid >> 5;
  int wm = w >> 2, wn = w & 3;
  int t0 = blockIdx.x * 128 + wm * 64;
  int n0 = blockIdx.y * 128 + wn * 32;
  int mrow = lane & 15;
  int kA = (lane >> 4) << 3;
  int kB = (lane >> 4) << 3;
  const bf16_t* pa[MW];
  const bf16_t* pb[NW];
#pragma unroll
  for (int mi = 0; mi < MW; ++mi) pa[mi] = A + (size_t)(t0 + mi * 16 + mrow) * lda + kA;
#pragma unroll
  for (int ni = 0; ni < NW; ++ni) pb[ni] = BT + (size_t)(n0 + ni * 16 + mrow) * ldb + kB;
  v8f acc[MW][NW] = {};
  v16bf a0[MW], b0[NW], a1[MW], b1[NW];
  load_frags(pa, pb, 0, a0, b0);
  for (int kk = 0; kk < K; kk += 64) {          // K/32 is even for all call sites
    load_frags(pa, pb, kk + 32, a1, b1);
    wmma8(a0, b0, acc);
    if (kk + 64 < K) load_frags(pa, pb, kk + 64, a0, b0);
    wmma8(a1, b1, acc);
  }
  int mlane = (lane >> 4) << 3, ncol = lane & 15;
#pragma unroll
  for (int mi = 0; mi < MW; ++mi)
#pragma unroll
    for (int ni = 0; ni < NW; ++ni)
#pragma unroll
      for (int r = 0; r < 8; ++r) {
        size_t t = (size_t)(t0 + mi * 16 + mlane + r);
        int n = n0 + ni * 16 + ncol;
        float v = acc[mi][ni][r];
        if (EPI == 0) {
          if (n < 512) out0[t * 512 + n] = v;
          else         out1[t * 512 + (n - 512)] = silu_f(v);
        } else if (EPI == 1) {
          if (n < 80) out0[t * 80 + n] = v;
        } else {
          out0[t * 256 + n] = v;
        }
      }
}

// conv1d(k=3) as im2col GEMM, K=768, reading halo-padded activations (no bounds checks)
__global__ void __launch_bounds__(256) k_conv_gemm(const bf16_t* __restrict__ xpad,
                                                   const bf16_t* __restrict__ WcbT,
                                                   const float* __restrict__ b_conv,
                                                   float* __restrict__ xt) {
  int tid = threadIdx.x, lane = tid & 31, w = tid >> 5;
  int wm = w >> 2, wn = w & 3;
  int t0 = blockIdx.x * 128 + wm * 64;
  int n0 = blockIdx.y * 128 + wn * 32;
  int mrow = lane & 15;
  int kA = (lane >> 4) << 3;
  int kB = (lane >> 4) << 3;
  const bf16_t* pa3[MW][3];
  const bf16_t* pb[NW];
#pragma unroll
  for (int mi = 0; mi < MW; ++mi) {
    int t = t0 + mi * 16 + mrow;
    int b = t >> 11, l = t & (LL - 1);
#pragma unroll
    for (int kc = 0; kc < 3; ++kc)
      pa3[mi][kc] = xpad + (size_t)(b * LP + l + kc) * 256 + kA;
  }
#pragma unroll
  for (int ni = 0; ni < NW; ++ni) pb[ni] = WcbT + (size_t)(n0 + ni * 16 + mrow) * 768 + kB;
  v8f acc[MW][NW] = {};
#pragma unroll
  for (int kc = 0; kc < 3; ++kc) {
    const bf16_t* pak[MW];
    const bf16_t* pbk[NW];
#pragma unroll
    for (int mi = 0; mi < MW; ++mi) pak[mi] = pa3[mi][kc];
#pragma unroll
    for (int ni = 0; ni < NW; ++ni) pbk[ni] = pb[ni] + kc * 256;
    v16bf a0[MW], b0[NW], a1[MW], b1[NW];
    load_frags(pak, pbk, 0, a0, b0);
    for (int kk = 0; kk < 256; kk += 64) {
      load_frags(pak, pbk, kk + 32, a1, b1);
      wmma8(a0, b0, acc);
      if (kk + 64 < 256) load_frags(pak, pbk, kk + 64, a0, b0);
      wmma8(a1, b1, acc);
    }
  }
  int mlane = (lane >> 4) << 3, ncol = lane & 15;
#pragma unroll
  for (int mi = 0; mi < MW; ++mi)
#pragma unroll
    for (int ni = 0; ni < NW; ++ni)
#pragma unroll
      for (int r = 0; r < 8; ++r) {
        size_t t = (size_t)(t0 + mi * 16 + mlane + r);
        int n = n0 + ni * 16 + ncol;
        xt[t * 256 + n] = silu_f(acc[mi][ni][r] + b_conv[n]);
      }
}

// block-wide layernorm over 256 values (256 threads)
__device__ __forceinline__ float ln256(float v, int d, const float* g, const float* bt,
                                       float* sv, float* red, float* sstat) {
  sv[d] = v;
  __syncthreads();
  if (d < 64) red[d] = sv[d] + sv[d + 64] + sv[d + 128] + sv[d + 192];
  __syncthreads();
  if (d < 32) red[d] += red[d + 32];
  __syncthreads();
  if (d == 0) { float s = 0.f; for (int i = 0; i < 32; ++i) s += red[i]; sstat[0] = s * (1.f / 256.f); }
  __syncthreads();
  float mu = sstat[0];
  float dv = v - mu;
  if (d < 64) {
    float a = sv[d] - mu, b2 = sv[d + 64] - mu, c = sv[d + 128] - mu, e = sv[d + 192] - mu;
    red[d] = a * a + b2 * b2 + c * c + e * e;
  }
  __syncthreads();
  if (d < 32) red[d] += red[d + 32];
  __syncthreads();
  if (d == 0) { float s = 0.f; for (int i = 0; i < 32; ++i) s += red[i]; sstat[1] = s * (1.f / 256.f); }
  __syncthreads();
  return dv * rsqrtf(sstat[1] + 1e-5f) * g[d] + bt[d];
}

// bf16 weight prep: all GEMM B operands stored transposed [N][K]; also zero conv halo rows
__global__ void k_prep(const float* W_conv, const float* W_inproj, const float* W_xproj,
                       const float* W_out, bf16_t* WcbT, bf16_t* WibT, bf16_t* WxbT,
                       bf16_t* WobT, bf16_t* xpad) {
  int i = blockIdx.x * 256 + threadIdx.x;
  if (i < 768 * 256) {                 // conv: [d][k*256+c] <- W_conv[d][c][k]
    int d = i / 768, rem = i % 768;
    int k = rem >> 8, c = rem & 255;
    WcbT[i] = (bf16_t)W_conv[(size_t)d * 768 + c * 3 + k];
  }
  if (i < 1024 * 256) {                // inproj: [n][k]
    int n = i >> 8, k = i & 255;
    WibT[i] = (bf16_t)W_inproj[(size_t)k * 1024 + n];
  }
  if (i < 128 * 512) {                 // xproj: [n][k], padded to 128 rows
    int n = i >> 9, k = i & 511;
    WxbT[i] = (n < 80) ? (bf16_t)W_xproj[(size_t)k * 80 + n] : (bf16_t)0.f;
  }
  if (i < 256 * 512) {                 // outproj: [n][k]
    int n = i >> 9, k = i & 511;
    WobT[i] = (bf16_t)W_out[(size_t)k * 256 + n];
  }
  if (i < Bb * 2 * 256) {              // conv halo rows (l = -1 and l = LL)
    int b = i >> 9, r2 = (i >> 8) & 1, d = i & 255;
    int row = r2 ? (LP - 1) : 0;
    xpad[(size_t)(b * LP + row) * 256 + d] = (bf16_t)0.f;
  }
}

// embedding + input proj + payload-LN mix (one block per token)
__global__ void k_embed_mix(const int* x_ids, const float* x_feats, const float* emb,
                            const float* W_in, const float* b_in, const float* W_pay,
                            const float* b_pay, const float* g_pn, const float* b_pn,
                            float* x0, bf16_t* xpad) {
  int t = blockIdx.x;
  int d = threadIdx.x;
  __shared__ float sin_[73];
  __shared__ float spay[128];
  __shared__ float red[64];
  __shared__ float sstat[2];
  int id = x_ids[t];
  const float* ft = x_feats + (size_t)t * 9;
  if (d < 64) sin_[d] = emb[(size_t)id * 64 + d];
  else if (d < 73) sin_[d] = ft[d - 64];
  __syncthreads();
  float xid = b_in[d];
  for (int j = 0; j < 73; ++j) xid += sin_[j] * W_in[j * 256 + d];
  if (d < 128) {
    float pv = b_pay[d];
    for (int j = 0; j < 9; ++j) pv += sin_[64 + j] * W_pay[j * 128 + d];
    spay[d] = pv;
  }
  __syncthreads();
  if (d < 64) red[d] = spay[d] + spay[d + 64];
  __syncthreads();
  if (d < 32) red[d] += red[d + 32];
  __syncthreads();
  if (d == 0) { float s = 0.f; for (int i = 0; i < 32; ++i) s += red[i]; sstat[0] = s * (1.f / 128.f); }
  __syncthreads();
  float mu = sstat[0];
  if (d < 64) {
    float a = spay[d] - mu, b2 = spay[d + 64] - mu;
    red[d] = a * a + b2 * b2;
  }
  __syncthreads();
  if (d < 32) red[d] += red[d + 32];
  __syncthreads();
  if (d == 0) { float s = 0.f; for (int i = 0; i < 32; ++i) s += red[i]; sstat[1] = s * (1.f / 128.f); }
  __syncthreads();
  int dp = d & 127;
  float payn = (spay[dp] - mu) * rsqrtf(sstat[1] + 1e-5f) * g_pn[dp] + b_pn[dp];
  float u = (id == 2047) ? 1.f : 0.f;
  float v = (1.f - u) * xid + u * payn;
  x0[(size_t)t * 256 + d] = v;
  int b = t >> 11, l = t & (LL - 1);
  xpad[(size_t)(b * LP + l + 1) * 256 + d] = (bf16_t)v;   // halo shift +1
}

// two-stage deterministic column reduction over L: stage 1 (grid Bb x 16, 128 rows each)
__global__ void k_psum(const float* src, float* partial) {
  int b = blockIdx.x, c = blockIdx.y, d = threadIdx.x;
  const float* base = src + ((size_t)b * LL + c * 128) * 256 + d;
  float s = 0.f;
  for (int l = 0; l < 128; ++l) s += base[(size_t)l * 256];
  partial[((size_t)b * 16 + c) * 256 + d] = s;
}
// stage 2 (grid Bb): combine 16 partials, scale
__global__ void k_psum2(const float* partial, float* dst, float scale) {
  int b = blockIdx.x, d = threadIdx.x;
  float s = 0.f;
#pragma unroll
  for (int c = 0; c < 16; ++c) s += partial[((size_t)b * 16 + c) * 256 + d];
  dst[b * 256 + d] = s * scale;
}

// ECA: conv k=5 over channels on pooled mean + sigmoid
__global__ void k_eca(const float* pooled, const float* w_eca, float* attn) {
  int b = blockIdx.x, d = threadIdx.x;
  float s = 0.f;
#pragma unroll
  for (int k = 0; k < 5; ++k) {
    int j = d + k - 2;
    if (j >= 0 && j < 256) s += w_eca[k] * (pooled[b * 256 + j] * (1.f / (float)LL));
  }
  attn[b * 256 + d] = 1.f / (1.f + expf(-s));
}

// x1 = LN(xt*attn + x0)
__global__ void k_ln1(const float* xt, const float* attn, const float* x0,
                      const float* g, const float* bt, float* x1, bf16_t* x1bf) {
  int t = blockIdx.x, d = threadIdx.x, b = t >> 11;
  __shared__ float sv[256], red[64], sstat[2];
  float v = xt[(size_t)t * 256 + d] * attn[b * 256 + d] + x0[(size_t)t * 256 + d];
  float o = ln256(v, d, g, bt, sv, red, sstat);
  x1[(size_t)t * 256 + d] = o;
  x1bf[(size_t)t * 256 + d] = (bf16_t)o;
}

// causal depthwise conv k=4 + silu
__global__ void k_dconv(const float* xmpre, const float* w_dconv, const float* b_dconv,
                        float* xm, bf16_t* xmbf) {
  int t = blockIdx.x, b = t >> 11, l = t & (LL - 1);
  for (int d = threadIdx.x; d < 512; d += 256) {
    float acc = b_dconv[d];
#pragma unroll
    for (int i = 0; i < 4; ++i) {
      int ll = l - 3 + i;
      if (ll >= 0) acc += w_dconv[d * 4 + i] * xmpre[((size_t)(b * LL + ll)) * 512 + d];
    }
    acc = silu_f(acc);
    xm[(size_t)t * 512 + d] = acc;
    xmbf[(size_t)t * 512 + d] = (bf16_t)acc;
  }
}

// delta = softplus(dt @ W_dt + b_dt)
__global__ void k_delta(const float* x_dbl, const float* W_dt, const float* b_dt, float* delta) {
  int t = blockIdx.x;
  __shared__ float sdt[16];
  if (threadIdx.x < 16) sdt[threadIdx.x] = x_dbl[(size_t)t * 80 + threadIdx.x];
  __syncthreads();
  for (int d = threadIdx.x; d < 512; d += 256) {
    float v = b_dt[d];
#pragma unroll
    for (int j = 0; j < 16; ++j) v += sdt[j] * W_dt[j * 512 + d];
    delta[(size_t)t * 512 + d] = (v > 20.f) ? v : log1pf(expf(v));
  }
}

// selective scan: wave per (b,d), lane = state. Fused epilogue (y + xm*D)*silu(z) -> bf16
__global__ void __launch_bounds__(1024) k_scan(
    const float* delta, const float* xm, const float* x_dbl, const float* sz,
    const float* A_log, const float* D_param, bf16_t* y2bf) {
  const int CH = 64;
  __shared__ float sdt[CH * 32], sxm[CH * 32], sB[CH * 32], sC[CH * 32], ssz[CH * 32], sy[CH * 32];
  int b  = blockIdx.x >> 4;
  int d0 = (blockIdx.x & 15) * 32;
  int tid = threadIdx.x;
  int lane = tid & 31;   // state index
  int w    = tid >> 5;   // channel within the 32-channel group
  int d = d0 + w;
  float Aval = -expf(A_log[d * 32 + lane]);
  float Dp = D_param[d];
  float h = 0.f;
  for (int l0 = 0; l0 < LL; l0 += CH) {
    __syncthreads();
#pragma unroll
    for (int i = 0; i < 2; ++i) {
      int e = i * 1024 + tid;
      int row = e >> 5, col = e & 31;
      size_t t = (size_t)(b * LL + l0 + row);
      sdt[e] = delta[t * 512 + d0 + col];
      sxm[e] = xm[t * 512 + d0 + col];
      ssz[e] = sz[t * 512 + d0 + col];
      sB[e]  = x_dbl[t * 80 + 16 + col];
      sC[e]  = x_dbl[t * 80 + 48 + col];
    }
    __syncthreads();
    for (int l = 0; l < CH; ++l) {
      float dt = sdt[l * 32 + w];
      float xv = sxm[l * 32 + w];
      h = __expf(dt * Aval) * h + (dt * xv) * sB[l * 32 + lane];
      float p = h * sC[l * 32 + lane];
      p += __shfl_xor(p, 16, 32);
      p += __shfl_xor(p, 8, 32);
      p += __shfl_xor(p, 4, 32);
      p += __shfl_xor(p, 2, 32);
      p += __shfl_xor(p, 1, 32);
      if (lane == 0) sy[l * 32 + w] = (p + xv * Dp) * ssz[l * 32 + w];
    }
    __syncthreads();
#pragma unroll
    for (int i = 0; i < 2; ++i) {
      int e = i * 1024 + tid;
      int row = e >> 5, col = e & 31;
      size_t t = (size_t)(b * LL + l0 + row);
      y2bf[t * 512 + d0 + col] = (bf16_t)sy[e];
    }
  }
}

// xfinal = LN(o2 + x1)
__global__ void k_ln2(const float* o2, const float* x1, const float* g, const float* bt,
                      float* xfinal) {
  int t = blockIdx.x, d = threadIdx.x;
  __shared__ float sv[256], red[64], sstat[2];
  float v = o2[(size_t)t * 256 + d] + x1[(size_t)t * 256 + d];
  xfinal[(size_t)t * 256 + d] = ln256(v, d, g, bt, sv, red, sstat);
}

// MLP head from precomputed mean (block per batch)
__global__ void k_head(const float* xmean_g, const float* W_c1, const float* b_c1,
                       const float* W_c2, const float* b_c2, float* out) {
  int b = blockIdx.x, d = threadIdx.x;
  __shared__ float xmean[256], hid[128];
  xmean[d] = xmean_g[b * 256 + d];
  __syncthreads();
  if (d < 128) {
    float a = b_c1[d];
    for (int j = 0; j < 256; ++j) a += xmean[j] * W_c1[j * 128 + d];
    hid[d] = silu_f(a);
  }
  __syncthreads();
  if (d < 2) {
    float a = b_c2[d];
    for (int j = 0; j < 128; ++j) a += hid[j] * W_c2[j * 2 + d];
    out[b * 2 + d] = a;
  }
}

// ---------------- launch ----------------
extern "C" void kernel_launch(void* const* d_in, const int* in_sizes, int n_in,
                              void* d_out, int out_size, void* d_ws, size_t ws_size,
                              hipStream_t stream) {
  const int*   x_ids    = (const int*)d_in[0];
  const float* x_feats  = (const float*)d_in[1];
  const float* emb      = (const float*)d_in[2];
  const float* W_in     = (const float*)d_in[3];
  const float* b_in     = (const float*)d_in[4];
  const float* W_pay    = (const float*)d_in[5];
  const float* b_pay    = (const float*)d_in[6];
  const float* g_pn     = (const float*)d_in[7];
  const float* b_pn     = (const float*)d_in[8];
  const float* W_conv   = (const float*)d_in[9];
  const float* b_conv   = (const float*)d_in[10];
  const float* g_n1     = (const float*)d_in[11];
  const float* b_n1     = (const float*)d_in[12];
  const float* w_eca    = (const float*)d_in[13];
  const float* W_inproj = (const float*)d_in[14];
  const float* w_dconv  = (const float*)d_in[15];
  const float* b_dconv  = (const float*)d_in[16];
  const float* W_xproj  = (const float*)d_in[17];
  const float* W_dt     = (const float*)d_in[18];
  const float* b_dt     = (const float*)d_in[19];
  const float* A_log    = (const float*)d_in[20];
  const float* D_param  = (const float*)d_in[21];
  const float* W_out    = (const float*)d_in[22];
  const float* g_n2     = (const float*)d_in[23];
  const float* b_n2     = (const float*)d_in[24];
  const float* W_c1     = (const float*)d_in[25];
  const float* b_c1     = (const float*)d_in[26];
  const float* W_c2     = (const float*)d_in[27];
  const float* b_c2     = (const float*)d_in[28];
  float* out = (float*)d_out;

  const size_t NT = (size_t)Bb * LL;  // 65536 tokens
  char* p = (char*)d_ws;
  auto alloc = [&](size_t bytes) -> void* {
    void* r = (void*)p;
    p += (bytes + 255) & ~(size_t)255;
    return r;
  };
  float*  x0      = (float*) alloc(NT * 256 * 4);
  bf16_t* xpad    = (bf16_t*)alloc((size_t)Bb * LP * 256 * 2);
  float*  xt      = (float*) alloc(NT * 256 * 4);
  float*  x1      = (float*) alloc(NT * 256 * 4);
  bf16_t* x1bf    = (bf16_t*)alloc(NT * 256 * 2);
  float*  xmpre   = (float*) alloc(NT * 512 * 4);
  float*  sz      = (float*) alloc(NT * 512 * 4);
  float*  xm      = (float*) alloc(NT * 512 * 4);
  bf16_t* xmbf    = (bf16_t*)alloc(NT * 512 * 2);
  float*  x_dbl   = (float*) alloc(NT * 80 * 4);
  bf16_t* y2bf    = (bf16_t*)alloc(NT * 512 * 2);
  float*  partial = (float*) alloc((size_t)Bb * 16 * 256 * 4);
  float*  pooled  = (float*) alloc(Bb * 256 * 4);
  float*  attn    = (float*) alloc(Bb * 256 * 4);
  float*  xmean   = (float*) alloc(Bb * 256 * 4);
  bf16_t* WcbT    = (bf16_t*)alloc(256 * 768 * 2);
  bf16_t* WibT    = (bf16_t*)alloc(1024 * 256 * 2);
  bf16_t* WxbT    = (bf16_t*)alloc(128 * 512 * 2);   // padded 80 -> 128 rows (zeros)
  bf16_t* WobT    = (bf16_t*)alloc(256 * 512 * 2);
  // lifetime-safe aliases (reduce peak workspace):
  float* delta  = xmpre;  // xmpre dead after k_dconv; delta written after
  float* o2     = x0;     // x0 dead after k_ln1
  float* xfinal = xt;     // xt dead after k_ln1 (pool of xt finished before too)

  k_prep<<<1024, 256, 0, stream>>>(W_conv, W_inproj, W_xproj, W_out,
                                   WcbT, WibT, WxbT, WobT, xpad);
  k_embed_mix<<<(int)NT, 256, 0, stream>>>(x_ids, x_feats, emb, W_in, b_in, W_pay, b_pay,
                                           g_pn, b_pn, x0, xpad);
  k_conv_gemm<<<dim3(512, 2), 256, 0, stream>>>(xpad, WcbT, b_conv, xt);
  k_psum<<<dim3(Bb, 16), 256, 0, stream>>>(xt, partial);
  k_psum2<<<Bb, 256, 0, stream>>>(partial, pooled, 1.f);
  k_eca<<<Bb, 256, 0, stream>>>(pooled, w_eca, attn);
  k_ln1<<<(int)NT, 256, 0, stream>>>(xt, attn, x0, g_n1, b_n1, x1, x1bf);
  k_gemm<0><<<dim3(512, 8), 256, 0, stream>>>(x1bf, 256, WibT, 256, 256, xmpre, sz);
  k_dconv<<<(int)NT, 256, 0, stream>>>(xmpre, w_dconv, b_dconv, xm, xmbf);
  k_gemm<1><<<dim3(512, 1), 256, 0, stream>>>(xmbf, 512, WxbT, 512, 512, x_dbl, nullptr);
  k_delta<<<(int)NT, 256, 0, stream>>>(x_dbl, W_dt, b_dt, delta);
  k_scan<<<Bb * 16, 1024, 0, stream>>>(delta, xm, x_dbl, sz, A_log, D_param, y2bf);
  k_gemm<2><<<dim3(512, 2), 256, 0, stream>>>(y2bf, 512, WobT, 512, 512, o2, nullptr);
  k_ln2<<<(int)NT, 256, 0, stream>>>(o2, x1, g_n2, b_n2, xfinal);
  k_psum<<<dim3(Bb, 16), 256, 0, stream>>>(xfinal, partial);
  k_psum2<<<Bb, 256, 0, stream>>>(partial, xmean, 1.f / (float)LL);
  k_head<<<Bb, 256, 0, stream>>>(xmean, W_c1, b_c1, W_c2, b_c2, out);
  (void)in_sizes; (void)n_in; (void)out_size; (void)ws_size;
}
